// PointPillarScatter_spa_63118839382092
// MI455X (gfx1250) — matrix operand order repaired and checked by reading, not compiled, not run on hardware
//
#include <hip/hip_runtime.h>
#include <hip/hip_fp16.h>
#include <hip/hip_bf16.h>

typedef _Float16 v8h  __attribute__((ext_vector_type(8)));
typedef _Float16 v16h __attribute__((ext_vector_type(16)));
typedef float    v8f  __attribute__((ext_vector_type(8)));

#define BB    2
#define PP    6000
#define NXc   432
#define NYc   496
#define HWc   (NXc*NYc)
#define CC    64
#define P2    6016              // keys padded to multiple of 32
#define TR    (BB*P2)           // 12032 total (padded) rows
#define TPBAT (PP/16)           // 375 query tiles per batch
#define NTIL  (TR/16)           // 752 padded row tiles
#define TILB  (P2/16)           // 376 padded tiles per batch

// ---- workspace layout (bytes) ----
constexpr size_t SZ_RAW = (size_t)TR*CC*4;  // f32 row buffer
constexpr size_t SZ_F16 = (size_t)TR*CC*2;  // f16 row buffer
constexpr size_t OFF_RAW = 0;
constexpr size_t OFF_ATT = OFF_RAW + SZ_RAW;
constexpr size_t OFF_N   = OFF_ATT + SZ_RAW;
constexpr size_t OFF_N2  = OFF_N   + SZ_F16;
constexpr size_t OFF_Q   = OFF_N2  + SZ_F16;
constexpr size_t OFF_K   = OFF_Q   + SZ_F16;
constexpr size_t OFF_VT  = OFF_K   + SZ_F16;
constexpr size_t OFF_WT  = OFF_VT  + SZ_F16;   // 5 x 64x64 f16 transposed weights

// ---------------- WMMA helpers (CDNA5 16x16x32 f16 -> f32) ----------------
__device__ __forceinline__ v8f wmma_f16(v16h a, v16h b, v8f c) {
  return __builtin_amdgcn_wmma_f32_16x16x32_f16(false, a, false, b, (short)0, c, false, false);
}

// A fragment (16x32 f16): lane<16 holds row=lane, K = {0..7,16..23}+32*kc
//                         lane>=16 holds row=lane-16, K = {8..15,24..31}+32*kc
__device__ __forceinline__ v16h ldA(const _Float16* base, int ld, int lane, int kc) {
  int m  = lane & 15;
  int b8 = (lane >> 4) << 3;
  const _Float16* p = base + (unsigned)(m * ld + kc * 32 + b8);
  union { v16h v; v8h h[2]; } u;
  u.h[0] = *(const v8h*)(p);
  u.h[1] = *(const v8h*)(p + 16);
  return u.v;
}

// B fragment (32x16 f16): lane%16 = N, K = {0..7,16..23} (lane<16) or {8..15,24..31} (lane>=16) + 32*kc
// base points at row-major [N][K] storage (i.e. B pre-transposed), ld = K stride
__device__ __forceinline__ v16h ldB(const _Float16* base, int ld, int lane, int kc) {
  int n  = lane & 15;
  int k0 = kc * 32 + ((lane >> 4) << 3);
  const _Float16* p = base + (unsigned)(n * ld + k0);
  union { v16h v; v8h h[2]; } u;
  u.h[0] = *(const v8h*)(p);
  u.h[1] = *(const v8h*)(p + 16);
  return u.v;
}

// ---------------- kernel 0: zero output ----------------
__global__ void __launch_bounds__(256) k_zero(float4* out, long n4) {
  long i = (long)blockIdx.x * blockDim.x + threadIdx.x;
  if (i < n4) { float4 z = {0.f, 0.f, 0.f, 0.f}; out[i] = z; }
}

// ---------------- kernel 1: convert 5 weights to f16, transposed ----------------
__global__ void __launch_bounds__(256) k_cvtw(const float* Wq, const float* Wk, const float* Wv,
                                              const float* W1, const float* W2, _Float16* Wt) {
  int t = blockIdx.x * blockDim.x + threadIdx.x;
  if (t >= 5 * 4096) return;
  int mat = t >> 12, nk = t & 4095, n = nk >> 6, k = nk & 63;
  const float* W = (mat == 0) ? Wq : (mat == 1) ? Wk : (mat == 2) ? Wv : (mat == 3) ? W1 : W2;
  Wt[t] = (_Float16)W[k * 64 + n];   // Wt[mat][n][k] = W[k][n]
}

// ---------------- kernel 2: per-pillar rotary feature + LayerNorm1 ----------------
__global__ void __launch_bounds__(128) k_prep(const float* pf, const int* coords,
                                              const float* g1, const float* be1,
                                              float* raw, _Float16* nrm) {
  int lane = threadIdx.x & 31;
  int row  = blockIdx.x * (blockDim.x >> 5) + (threadIdx.x >> 5);
  if (row >= TR) return;
  int b = (row >= P2) ? 1 : 0;          // BB == 2
  int p = row - b * P2;
  float r1 = 0.f, r2 = 0.f;
  if (p < PP) {
    int gp = b * PP + p;
    float x1 = pf[(size_t)gp * CC + lane];
    float x2 = pf[(size_t)gp * CC + 32 + lane];
    int y = coords[gp * 4 + 2];
    int x = coords[gp * 4 + 3];
    float hy = -1.f + 2.f * (float)y / (float)(NYc - 1);
    float wx = -1.f + 2.f * (float)x / (float)(NXc - 1);
    float th = __expf(-(float)lane * (9.210340371976184f / 32.f));  // 10000^(-k/32)
    float A  = __cosf(hy * th) * __cosf(wx * th);
    float Bt = __sinf(hy * th) * __sinf(wx * th);
    r1 = x1 * (1.f + A) - x2 * Bt;
    r2 = x1 * Bt + x2 * (1.f + A);
  }
  float s = r1 + r2;
  #pragma unroll
  for (int m = 16; m; m >>= 1) s += __shfl_xor(s, m, 32);
  float mean = s * (1.f / 64.f);
  float d1 = r1 - mean, d2 = r2 - mean;
  float v = d1 * d1 + d2 * d2;
  #pragma unroll
  for (int m = 16; m; m >>= 1) v += __shfl_xor(v, m, 32);
  float rs = rsqrtf(v * (1.f / 64.f) + 1e-5f);
  raw[(size_t)row * CC + lane]      = r1;
  raw[(size_t)row * CC + 32 + lane] = r2;
  nrm[(size_t)row * CC + lane]      = (_Float16)(d1 * rs * g1[lane] + be1[lane]);
  nrm[(size_t)row * CC + 32 + lane] = (_Float16)(d2 * rs * g1[32 + lane] + be1[32 + lane]);
}

// ---------------- kernel 3: QKV GEMM (WMMA), V stored transposed ----------------
__global__ void __launch_bounds__(128) k_qkv(const _Float16* nrm, const _Float16* Wt,
                                             const float* bq, const float* bk, const float* bv,
                                             _Float16* Q, _Float16* K, _Float16* Vt) {
  int lane = threadIdx.x & 31;
  int w = blockIdx.x * (blockDim.x >> 5) + (threadIdx.x >> 5);
  if (w >= NTIL * 3) return;
  int mat = (w >= 2 * NTIL) ? 2 : (w >= NTIL) ? 1 : 0;   // wave-uniform, scalar branch
  int tile = w - mat * NTIL;
  int rbase = tile * 16;
  v16h a0 = ldA(nrm + (size_t)rbase * CC, CC, lane, 0);
  v16h a1 = ldA(nrm + (size_t)rbase * CC, CC, lane, 1);
  const _Float16* wb = Wt + mat * 4096;
  int n16 = lane & 15, mh = (lane >> 4) << 3;
  if (mat < 2) {
    _Float16* outm = (mat == 0) ? Q : K;
    const float* bias = (mat == 0) ? bq : bk;
    #pragma unroll
    for (int t = 0; t < 4; t++) {
      v8f c = {};
      c = wmma_f16(a0, ldB(wb + t * 16 * CC, CC, lane, 0), c);
      c = wmma_f16(a1, ldB(wb + t * 16 * CC, CC, lane, 1), c);
      int col = t * 16 + n16;
      float bc = bias[col];
      #pragma unroll
      for (int r = 0; r < 8; r++)
        outm[(size_t)(rbase + r + mh) * CC + col] = (_Float16)(c[r] + bc);
    }
  } else {
    int bb = (tile >= TILB) ? 1 : 0;           // tiles never straddle batches
    int rb0 = (tile - bb * TILB) * 16;
    _Float16* vtb = Vt + (size_t)bb * CC * P2;
    #pragma unroll
    for (int t = 0; t < 4; t++) {
      v8f c = {};
      c = wmma_f16(a0, ldB(wb + t * 16 * CC, CC, lane, 0), c);
      c = wmma_f16(a1, ldB(wb + t * 16 * CC, CC, lane, 1), c);
      int col = t * 16 + n16;
      float bc = bv[col];
      #pragma unroll
      for (int r = 0; r < 8; r++)
        vtb[(size_t)col * P2 + (rb0 + r + mh)] = (_Float16)(c[r] + bc);  // V^T: [col][row]
    }
  }
}

// ---------------- kernel 4: flash attention + residual + LayerNorm2 ----------------
__global__ void __launch_bounds__(128) k_flash(const _Float16* Q, const _Float16* K,
                                               const _Float16* Vt, const float* raw,
                                               const float* g2, const float* be2,
                                               float* att, _Float16* n2) {
  __shared__ _Float16 lsP[4][16 * 32];
  int lane = threadIdx.x & 31;
  int wib  = threadIdx.x >> 5;
  int w = blockIdx.x * (blockDim.x >> 5) + wib;
  if (w >= BB * TPBAT) return;
  int b = (w >= TPBAT) ? 1 : 0;
  int tt = w - b * TPBAT;
  size_t qrow = (size_t)b * P2 + (size_t)tt * 16;
  v16h qa0 = ldA(Q + qrow * CC, CC, lane, 0);
  v16h qa1 = ldA(Q + qrow * CC, CC, lane, 1);
  v8f acc0 = {}, acc1 = {}, acc2 = {}, acc3 = {};
  v8f mi, li;
  #pragma unroll
  for (int r = 0; r < 8; r++) { mi[r] = -1e30f; li[r] = 0.f; }
  int n16 = lane & 15, mh = (lane >> 4) << 3;
  _Float16* lp = lsP[wib];
  const int NC = (PP + 31) / 32;                // 188 key chunks of 32
  for (int kc = 0; kc < NC; ++kc) {
    int kb = kc * 32;
    const _Float16* kp = K + ((size_t)b * P2 + kb) * CC;
    v8f s0 = {}, s1 = {};
    s0 = wmma_f16(qa0, ldB(kp, CC, lane, 0), s0);
    s0 = wmma_f16(qa1, ldB(kp, CC, lane, 1), s0);
    s1 = wmma_f16(qa0, ldB(kp + 16 * CC, CC, lane, 0), s1);
    s1 = wmma_f16(qa1, ldB(kp + 16 * CC, CC, lane, 1), s1);
    bool inv0 = (kb + n16 >= PP);
    bool inv1 = (kb + 16 + n16 >= PP);
    #pragma unroll
    for (int r = 0; r < 8; r++) {
      if (inv0) s0[r] = -1e30f;
      if (inv1) s1[r] = -1e30f;
    }
    v8f mn, al, p0, p1;
    #pragma unroll
    for (int r = 0; r < 8; r++) {
      float mx = fmaxf(s0[r], s1[r]);
      #pragma unroll
      for (int m = 8; m; m >>= 1) mx = fmaxf(mx, __shfl_xor(mx, m, 32));
      mn[r] = fmaxf(mi[r], mx);
      al[r] = __expf(mi[r] - mn[r]);
      p0[r] = __expf(s0[r] - mn[r]);
      p1[r] = __expf(s1[r] - mn[r]);
      float ls = p0[r] + p1[r];
      #pragma unroll
      for (int m = 8; m; m >>= 1) ls += __shfl_xor(ls, m, 32);
      li[r] = li[r] * al[r] + ls;
      mi[r] = mn[r];
    }
    #pragma unroll
    for (int r = 0; r < 8; r++) {
      acc0[r] *= al[r]; acc1[r] *= al[r]; acc2[r] *= al[r]; acc3[r] *= al[r];
      lp[(r + mh) * 32 + n16]      = (_Float16)p0[r];   // transpose P via LDS
      lp[(r + mh) * 32 + 16 + n16] = (_Float16)p1[r];
    }
    asm volatile("s_wait_dscnt 0" ::: "memory");
    union { v16h v; v8h h[2]; } pa;
    pa.h[0] = *(const v8h*)&lp[n16 * 32 + mh];
    pa.h[1] = *(const v8h*)&lp[n16 * 32 + mh + 16];
    const _Float16* vb = Vt + (size_t)b * CC * P2 + kb;
    acc0 = wmma_f16(pa.v, ldB(vb + (size_t)0 * 16 * P2, P2, lane, 0), acc0);
    acc1 = wmma_f16(pa.v, ldB(vb + (size_t)1 * 16 * P2, P2, lane, 0), acc1);
    acc2 = wmma_f16(pa.v, ldB(vb + (size_t)2 * 16 * P2, P2, lane, 0), acc2);
    acc3 = wmma_f16(pa.v, ldB(vb + (size_t)3 * 16 * P2, P2, lane, 0), acc3);
  }
  // epilogue: 1/l scaling, residual, LayerNorm2
  float g2v[4], be2v[4];
  #pragma unroll
  for (int t = 0; t < 4; t++) { g2v[t] = g2[t * 16 + n16]; be2v[t] = be2[t * 16 + n16]; }
  #pragma unroll
  for (int r = 0; r < 8; r++) {
    float inv = 1.f / li[r];
    size_t row = qrow + r + mh;
    float o[4];
    o[0] = acc0[r] * inv + raw[row * CC +  0 + n16];
    o[1] = acc1[r] * inv + raw[row * CC + 16 + n16];
    o[2] = acc2[r] * inv + raw[row * CC + 32 + n16];
    o[3] = acc3[r] * inv + raw[row * CC + 48 + n16];
    float s = o[0] + o[1] + o[2] + o[3];
    #pragma unroll
    for (int m = 8; m; m >>= 1) s += __shfl_xor(s, m, 32);
    float mean = s * (1.f / 64.f);
    float vv = 0.f;
    #pragma unroll
    for (int t = 0; t < 4; t++) { float d = o[t] - mean; vv += d * d; }
    #pragma unroll
    for (int m = 8; m; m >>= 1) vv += __shfl_xor(vv, m, 32);
    float rs = rsqrtf(vv * (1.f / 64.f) + 1e-5f);
    #pragma unroll
    for (int t = 0; t < 4; t++) {
      att[row * CC + t * 16 + n16] = o[t];
      n2[row * CC + t * 16 + n16] = (_Float16)((o[t] - mean) * rs * g2v[t] + be2v[t]);
    }
  }
}

// ---------------- kernel 5: FFN (GELU-erf) + residual, fused scatter to output ----------------
__global__ void __launch_bounds__(128) k_ffn(const _Float16* n2, const _Float16* Wt,
                                             const float* b1, const float* b2,
                                             const float* att, const int* coords, float* out) {
  __shared__ _Float16 lsH[4][16 * 64];
  int lane = threadIdx.x & 31;
  int wib  = threadIdx.x >> 5;
  int w = blockIdx.x * (blockDim.x >> 5) + wib;
  if (w >= BB * TPBAT) return;
  int b = (w >= TPBAT) ? 1 : 0;
  int tt = w - b * TPBAT;
  size_t row0 = (size_t)b * P2 + (size_t)tt * 16;
  v16h a0 = ldA(n2 + row0 * CC, CC, lane, 0);
  v16h a1 = ldA(n2 + row0 * CC, CC, lane, 1);
  const _Float16* W1t = Wt + 3 * 4096;
  const _Float16* W2t = Wt + 4 * 4096;
  _Float16* lp = lsH[wib];
  int n16 = lane & 15, mh = (lane >> 4) << 3;
  #pragma unroll
  for (int t = 0; t < 4; t++) {
    v8f c = {};
    c = wmma_f16(a0, ldB(W1t + t * 16 * CC, CC, lane, 0), c);
    c = wmma_f16(a1, ldB(W1t + t * 16 * CC, CC, lane, 1), c);
    int col = t * 16 + n16;
    float bc = b1[col];
    #pragma unroll
    for (int r = 0; r < 8; r++) {
      float x = c[r] + bc;
      float g = 0.5f * x * (1.f + erff(x * 0.7071067811865476f));  // exact GELU
      lp[(r + mh) * CC + col] = (_Float16)g;
    }
  }
  asm volatile("s_wait_dscnt 0" ::: "memory");
  union { v16h v; v8h h[2]; } h0, h1;
  h0.h[0] = *(const v8h*)&lp[n16 * CC + mh];
  h0.h[1] = *(const v8h*)&lp[n16 * CC + mh + 16];
  h1.h[0] = *(const v8h*)&lp[n16 * CC + 32 + mh];
  h1.h[1] = *(const v8h*)&lp[n16 * CC + 32 + mh + 16];
  #pragma unroll
  for (int t = 0; t < 4; t++) {
    v8f c = {};
    c = wmma_f16(h0.v, ldB(W2t + t * 16 * CC, CC, lane, 0), c);
    c = wmma_f16(h1.v, ldB(W2t + t * 16 * CC, CC, lane, 1), c);
    int col = t * 16 + n16;
    float bc = b2[col];
    #pragma unroll
    for (int r = 0; r < 8; r++) {
      size_t row = row0 + r + mh;
      float u = c[r] + bc + att[row * CC + col];
      int rb = tt * 16 + r + mh;           // row within batch (< 6000)
      int gp = b * PP + rb;
      int y = coords[gp * 4 + 2];
      int x = coords[gp * 4 + 3];
      out[((size_t)b * CC + col) * HWc + (size_t)((unsigned)(y * NXc + x))] = u;
    }
  }
}

// ---------------- launch ----------------
extern "C" void kernel_launch(void* const* d_in, const int* in_sizes, int n_in,
                              void* d_out, int out_size, void* d_ws, size_t ws_size,
                              hipStream_t stream) {
  const float* pf   = (const float*)d_in[0];
  const int* coords = (const int*)d_in[1];
  const float* Wq = (const float*)d_in[2];  const float* bq = (const float*)d_in[3];
  const float* Wk = (const float*)d_in[4];  const float* bk = (const float*)d_in[5];
  const float* Wv = (const float*)d_in[6];  const float* bv = (const float*)d_in[7];
  const float* W1 = (const float*)d_in[8];  const float* b1 = (const float*)d_in[9];
  const float* W2 = (const float*)d_in[10]; const float* b2 = (const float*)d_in[11];
  const float* g1 = (const float*)d_in[12]; const float* be1 = (const float*)d_in[13];
  const float* g2 = (const float*)d_in[14]; const float* be2 = (const float*)d_in[15];

  char* ws = (char*)d_ws;
  float*     raw = (float*)(ws + OFF_RAW);
  float*     att = (float*)(ws + OFF_ATT);
  _Float16*  nrm = (_Float16*)(ws + OFF_N);
  _Float16*  n2b = (_Float16*)(ws + OFF_N2);
  _Float16*  Qb  = (_Float16*)(ws + OFF_Q);
  _Float16*  Kb  = (_Float16*)(ws + OFF_K);
  _Float16*  Vt  = (_Float16*)(ws + OFF_VT);
  _Float16*  Wt  = (_Float16*)(ws + OFF_WT);
  float* outp = (float*)d_out;

  long n4 = (long)BB * CC * HWc / 4;
  k_zero<<<(int)((n4 + 255) / 256), 256, 0, stream>>>((float4*)outp, n4);
  k_cvtw<<<(5 * 4096 + 255) / 256, 256, 0, stream>>>(Wq, Wk, Wv, W1, W2, Wt);
  k_prep<<<TR / 4, 128, 0, stream>>>(pf, coords, g1, be1, raw, nrm);
  k_qkv<<<(NTIL * 3) / 4, 128, 0, stream>>>(nrm, Wt, bq, bk, bv, Qb, Kb, Vt);
  k_flash<<<(BB * TPBAT + 3) / 4, 128, 0, stream>>>(Qb, Kb, Vt, raw, g2, be2, att, n2b);
  k_ffn<<<(BB * TPBAT + 3) / 4, 128, 0, stream>>>(n2b, Wt, b1, b2, att, coords, outp);
}
